// NeuralNetwork_19086834663865
// MI455X (gfx1250) — matrix operand order repaired and checked by reading, not compiled, not run on hardware
//
#include <hip/hip_runtime.h>
#include <hip/hip_bf16.h>

// CDNA5 (gfx1250) deep-hedging Monte Carlo: MLP via v_wmma_f32_16x16x32_f16.
// - batch = N dim (16 paths/tile), neurons = M (30-wide layers -> 2 tiles),
//   features = K padded to 32, bias folded in as an extra K row.
// - each wave32 carries TWO independent 16-path tiles whose MLP evals are
//   interleaved to fill the f16-WMMA RAW hazard windows and swizzle waits.
// - D->B relayout packs f32 pairs to f16 FIRST, then exchanges packed dwords
//   lane<->lane^16 via ds_swizzle (half the LDS-pipe traffic).
// - IMPORTANT: all ds_swizzles are issued unconditionally (full EXEC) BEFORE
//   any lane-divergent select; putting swzi() inside a divergent ternary makes
//   clang predicate the swizzle under partial EXEC, which reads zeros from the
//   disabled partner lanes (and bloats SALU with saveexec pairs).
//
// Roofline: ~75 GFLOP useful vs ~100 MB DRAM (4.3us floor @ 23.3 TB/s) ->
// compute-bound; f16 WMMA is the MI455X fast path.

typedef __attribute__((ext_vector_type(16))) _Float16 v16h;
typedef __attribute__((ext_vector_type(2)))  _Float16 h2;
typedef __attribute__((ext_vector_type(8)))  float    v8f;
typedef __attribute__((ext_vector_type(8)))  int      v8i;

// lane <-> lane^16 exchange of a packed dword: ds_swizzle xor=0x10, and=0x1f
static __device__ __forceinline__ int swzi(int i) {
    return __builtin_amdgcn_ds_swizzle(i, 0x401f);
}

static __device__ __forceinline__ float swzf(float x) {
    return __builtin_bit_cast(float, swzi(__builtin_bit_cast(int, x)));
}

// pack two f32 into one dword of two f16 (emits v_cvt_pk_f16_f32)
static __device__ __forceinline__ int pkh(float lo, float hi) {
    h2 v;
    v[0] = (_Float16)lo;
    v[1] = (_Float16)hi;
    return __builtin_bit_cast(int, v);
}

static __device__ __forceinline__ v8f wmma16(v16h a, v16h b, v8f c) {
    // (neg_a, A, neg_b, B, c_mod, C, reuse_a, reuse_b)
    return __builtin_amdgcn_wmma_f32_16x16x32_f16(false, a, false, b,
                                                  (short)0, c, false, false);
}

// Build A fragment = W^T for one 16-row M tile, bias folded at K==FIN.
// A 16x32 fragment, contiguous K order used here (same as B): lane (r,g): half h -> K = 16g+h, M=r+16*TILE
template<int FIN, int FOUT, int TILE>
static __device__ __forceinline__ v16h loadAfrag(const float* __restrict__ W,
                                                 const float* __restrict__ bias,
                                                 int r, int g) {
    v16h a;
#pragma unroll
    for (int h = 0; h < 16; ++h) {
        int k = 16 * g + h;
        int m = TILE * 16 + r;
        float v = 0.0f;
        if (m < FOUT) {
            if (k < FIN)       v = W[k * FOUT + m];
            else if (k == FIN) v = bias[m];
        }
        a[h] = (_Float16)v;
    }
    return a;
}

// B fragment from two D tiles (30 valid rows), relu fused, packed exchange.
// B 32x16 fragment, contiguous K order used here (same as A): lane (p,g): dword v -> K = 16g+2v, 16g+2v+1; N = p.
// K==30 bias-1: dword7 on g1 lanes = {1.0h, 0h} = 0x3c00 (rows 30/31 are 0).
static __device__ __forceinline__ v16h buildB30(const v8f& c0, const v8f& c1, bool g0) {
    int p0[4], p1[4], sw0[4], sw1[4];
#pragma unroll
    for (int v = 0; v < 4; ++v) {
        p0[v] = pkh(fmaxf(c0[2 * v], 0.0f), fmaxf(c0[2 * v + 1], 0.0f));
        p1[v] = pkh(fmaxf(c1[2 * v], 0.0f), fmaxf(c1[2 * v + 1], 0.0f));
    }
    // Unconditional exchanges (full EXEC) before any divergent select.
#pragma unroll
    for (int v = 0; v < 4; ++v) {
        sw0[v] = swzi(p0[v]);
        sw1[v] = swzi(p1[v]);
    }
    v8i bi;
#pragma unroll
    for (int v = 0; v < 4; ++v)            // K=2v,2v+1 (g0) / 16+2v,+1 (g1)
        bi[v] = g0 ? p0[v] : sw1[v];
#pragma unroll
    for (int v = 4; v < 7; ++v)            // K=2v,2v+1 (g0) / 16+2v,+1 (g1)
        bi[v] = g0 ? sw0[v - 4] : p1[v - 4];
    bi[7] = g0 ? sw0[3] : 0x3c00;          // g1: K=30 -> 1.0 (bias), K=31 -> 0
    return __builtin_bit_cast(v16h, bi);
}

// B fragment from a 5-valid-row D tile (rows 5..15 exactly 0, no relu).
// g0: K=0..4 data, K=5 = 1.0 (bias); g1: all zero. No shuffles needed.
static __device__ __forceinline__ v16h buildB5(const v8f& d, bool g0) {
    int q0 = pkh(d[0], d[1]);
    int q1 = pkh(d[2], d[3]);
    int q2 = pkh(d[4], 1.0f);
    v8i bi;
    bi[0] = g0 ? q0 : 0;
    bi[1] = g0 ? q1 : 0;
    bi[2] = g0 ? q2 : 0;
#pragma unroll
    for (int v = 3; v < 8; ++v) bi[v] = 0;
    return __builtin_bit_cast(v16h, bi);
}

// Layer-1 input: [price, t, T-t, log(price/K), 1, 0...] on g0 lanes.
static __device__ __forceinline__ v16h buildBx(float price, float t, float Tm,
                                               float lg, bool g0) {
    int q0 = pkh(price, t);
    int q1 = pkh(Tm, lg);
    v8i bi;
    bi[0] = g0 ? q0 : 0;
    bi[1] = g0 ? q1 : 0;
    bi[2] = g0 ? 0x3c00 : 0;               // K=4 -> 1.0 (bias row)
#pragma unroll
    for (int v = 3; v < 8; ++v) bi[v] = 0;
    return __builtin_bit_cast(v16h, bi);
}

// Two interleaved MLP evaluations (independent tiles A/B) sharing weights.
static __device__ __forceinline__ void mlp2(const v16h (&aw)[9], v16h bxA, v16h bxB,
                                            bool g0, v8f& outA, v8f& outB) {
    v8f z = {};
    v8f a0 = wmma16(aw[0], bxA, z);
    v8f b0 = wmma16(aw[0], bxB, z);
    v8f a1 = wmma16(aw[1], bxA, z);
    v8f b1 = wmma16(aw[1], bxB, z);
    v16h fA = buildB30(a0, a1, g0);
    v16h fB = buildB30(b0, b1, g0);
    v8f dA = wmma16(aw[2], fA, z);          // 30 -> 5
    v8f dB = wmma16(aw[2], fB, z);
    fA = buildB5(dA, g0);
    fB = buildB5(dB, g0);
    a0 = wmma16(aw[3], fA, z);
    b0 = wmma16(aw[3], fB, z);
    a1 = wmma16(aw[4], fA, z);
    b1 = wmma16(aw[4], fB, z);
    fA = buildB30(a0, a1, g0);
    fB = buildB30(b0, b1, g0);
    dA = wmma16(aw[5], fA, z);              // 30 -> 5
    dB = wmma16(aw[5], fB, z);
    fA = buildB5(dA, g0);
    fB = buildB5(dB, g0);
    a0 = wmma16(aw[6], fA, z);
    b0 = wmma16(aw[6], fB, z);
    a1 = wmma16(aw[7], fA, z);
    b1 = wmma16(aw[7], fB, z);
    fA = buildB30(a0, a1, g0);
    fB = buildB30(b0, b1, g0);
    outA = wmma16(aw[8], fA, z);            // 30 -> 2
    outB = wmma16(aw[8], fB, z);
}

__global__ __launch_bounds__(256) void deephedge_wmma_kernel(
    const float* __restrict__ S0, const float* __restrict__ Karr,
    const float* __restrict__ Tarr, const float* __restrict__ BM,
    const float* __restrict__ W1, const float* __restrict__ b1,
    const float* __restrict__ W2, const float* __restrict__ b2,
    const float* __restrict__ W3, const float* __restrict__ b3,
    const float* __restrict__ W4, const float* __restrict__ b4,
    const float* __restrict__ W5, const float* __restrict__ b5,
    const float* __restrict__ W6, const float* __restrict__ b6,
    float* __restrict__ out, int nPaths)
{
    const int lane = threadIdx.x & 31;
    const int wave = blockIdx.x * (blockDim.x >> 5) + (threadIdx.x >> 5);
    const int r = lane & 15;
    const int g = lane >> 4;
    const bool g0 = (g == 0);
    const int pA = wave * 32 + r;           // tile A path (replicated on lane pair)
    const int pB = pA + 16;                 // tile B path

    // Weight fragments (VGPR-resident, shared by both tiles, all 40 evals).
    v16h aw[9];
    aw[0] = loadAfrag<4, 30, 0>(W1, b1, r, g);
    aw[1] = loadAfrag<4, 30, 1>(W1, b1, r, g);
    aw[2] = loadAfrag<30, 5, 0>(W2, b2, r, g);
    aw[3] = loadAfrag<5, 30, 0>(W3, b3, r, g);
    aw[4] = loadAfrag<5, 30, 1>(W3, b3, r, g);
    aw[5] = loadAfrag<30, 5, 0>(W4, b4, r, g);
    aw[6] = loadAfrag<5, 30, 0>(W5, b5, r, g);
    aw[7] = loadAfrag<5, 30, 1>(W5, b5, r, g);
    aw[8] = loadAfrag<30, 2, 0>(W6, b6, r, g);

    float priceA = S0[pA],  priceB = S0[pB];
    float KvA    = Karr[pA], KvB   = Karr[pB];
    float TvA    = Tarr[pA], TvB   = Tarr[pB];
    float hedgeA = 0.0f,    hedgeB = 0.0f;

    const float* bmA = BM + (size_t)pA * 20;
    const float* bmB = BM + (size_t)pB * 20;

    for (int s = 0; s < 20; ++s) {
        float dWa = bmA[s];
        float dWb = bmB[s];
        float time  = (float)(s + 1) * 0.05f;     // (idx+1)/N
        float incA  = dWa * 0.05f;                // dW * STEP_SIZE
        float incB  = dWb * 0.05f;
        float compA = (time <= TvA) ? 1.0f : 0.0f;
        float compB = (time <= TvB) ? 1.0f : 0.0f;

        // eval 1: leverage from pre-update price
        v16h bxA = buildBx(priceA, time, TvA - time, __logf(priceA / KvA), g0);
        v16h bxB = buildBx(priceB, time, TvB - time, __logf(priceB / KvB), g0);
        v8f dA, dB;
        mlp2(aw, bxA, bxB, g0, dA, dB);
        // broadcast row0 -> both lane halves: swizzle FIRST (full EXEC), then select
        float sA0 = swzf(dA[0]);
        float sB0 = swzf(dB[0]);
        float levA = g0 ? dA[0] : sA0;
        float levB = g0 ? dB[0] : sB0;
        priceA = priceA + compA * levA * priceA * incA;
        priceB = priceB + compB * levB * priceB * incB;

        // eval 2: hedge ratio from post-update price
        bxA = buildBx(priceA, time, TvA - time, __logf(priceA / KvA), g0);
        bxB = buildBx(priceB, time, TvB - time, __logf(priceB / KvB), g0);
        mlp2(aw, bxA, bxB, g0, dA, dB);
        float sA1 = swzf(dA[1]);
        float sB1 = swzf(dB[1]);
        float hedA = g0 ? dA[1] : sA1;
        float hedB = g0 ? dB[1] : sB1;
        hedgeA = hedgeA + compA * hedA * incA;
        hedgeB = hedgeB + compB * hedB * incB;
    }

    float resA = fmaxf(priceA - KvA, 0.0f) - hedgeA;
    float resB = fmaxf(priceB - KvB, 0.0f) - hedgeB;
    if (g0 && pA < nPaths) out[pA] = resA;
    if (g0 && pB < nPaths) out[pB] = resB;
}

extern "C" void kernel_launch(void* const* d_in, const int* in_sizes, int n_in,
                              void* d_out, int out_size, void* d_ws, size_t ws_size,
                              hipStream_t stream) {
    const float* S0 = (const float*)d_in[0];
    const float* K  = (const float*)d_in[1];
    const float* T  = (const float*)d_in[2];
    const float* BM = (const float*)d_in[3];
    const float* W1 = (const float*)d_in[4];  const float* b1 = (const float*)d_in[5];
    const float* W2 = (const float*)d_in[6];  const float* b2 = (const float*)d_in[7];
    const float* W3 = (const float*)d_in[8];  const float* b3 = (const float*)d_in[9];
    const float* W4 = (const float*)d_in[10]; const float* b4 = (const float*)d_in[11];
    const float* W5 = (const float*)d_in[12]; const float* b5 = (const float*)d_in[13];
    const float* W6 = (const float*)d_in[14]; const float* b6 = (const float*)d_in[15];
    float* out = (float*)d_out;

    int nPaths = in_sizes[0];
    // 256 threads = 8 waves = 256 paths per block (32 paths per wave)
    int blocks = (nPaths + 255) / 256;
    deephedge_wmma_kernel<<<blocks, 256, 0, stream>>>(
        S0, K, T, BM, W1, b1, W2, b2, W3, b3, W4, b4, W5, b5, W6, b6,
        out, nPaths);
}
